// ModelMultitaskBinary_14139032338491
// MI455X (gfx1250) — matrix operand order repaired and checked by reading, not compiled, not run on hardware
//
#include <hip/hip_runtime.h>
#include <hip/hip_bf16.h>

// ---------------------------------------------------------------------------
// MMoE multitask model on MI455X (gfx1250), bf16 WMMA (16x16x32) with f32 acc.
// Pipeline:
//   1. transpose_cvt: fp32 weights -> bf16 [out,in] so WMMA A/B tiles load as
//      contiguous 16B/32B per-lane chunks (ISA 7.12.2 layouts).
//      wgate_cvt: w_gate [T,H,E] -> bf16 [T*E][H] for vectorized gating dots.
//   2. gemm_bias_act: h1 = relu(x@W1+b1); h = h1@W2+b2   (bf16 out)
//   3. moe_expert_kernel: per 16-token tile: fp32-accum gating (top-3 softmax
//      in LDS), loop 6 experts with eh staged in LDS, eo + gate-combine in
//      regs, write mo bf16 [T,N,H]. Dense eo (402MB) never materialized.
//   4. tower_kernel: th = relu(mo@tw1+tb1) with tw2-dot fused in epilogue via
//      shfl_xor reduction + f32 atomics into out (pre-init with tb2).
// No software prefetch: __builtin_prefetch only reaches SE/DEV scope (no WGP$
// fill per ISA) and its live address registers caused inner-loop VGPR spills.
// ---------------------------------------------------------------------------

typedef __bf16 bf16;
typedef __attribute__((ext_vector_type(16))) __bf16 v16bf;
typedef __attribute__((ext_vector_type(8)))  float  v8f;

#define NTOK  16384   // B*C
#define DIMH  1024    // D == H == EH == TH
#define NEXP  6
#define NTASK 3

__device__ __forceinline__ v8f vzero8() {
  v8f v = {0.f, 0.f, 0.f, 0.f, 0.f, 0.f, 0.f, 0.f};
  return v;
}

__device__ __forceinline__ v8f wmma_bf16(v16bf a, v16bf b, v8f c) {
  // D(f32 16x16) = A(16x32 bf16) * B(32x16 bf16) + C
  return __builtin_amdgcn_wmma_f32_16x16x32_bf16(false, a, false, b,
                                                 (short)0, c, false, false);
}

// A tile 16(M)x32(K) from row-major [*, ld] bf16.
// ISA layout: lanes hold M=lane&15; elems 0-7 -> K = (lane>>4)*8 + 0..7,
// elems 8-15 -> K = 16 + (lane>>4)*8 + 0..7  => two 16B loads per lane.
__device__ __forceinline__ v16bf load_tileA_bf16(const bf16* base, int ld,
                                                 int row0, int k0) {
  int lane = threadIdx.x & 31;
  const bf16* p = base + (size_t)(row0 + (lane & 15)) * ld + k0 + (lane >> 4) * 8;
  union { v16bf v; uint4 q[2]; } u;
  u.q[0] = *(const uint4*)p;
  u.q[1] = *(const uint4*)(p + 16);
  return u.v;
}

// Same A layout, but source is fp32 (converted on the fly, RNE).
__device__ __forceinline__ v16bf load_tileA_f32(const float* base, int ld,
                                                int row0, int k0) {
  int lane = threadIdx.x & 31;
  const float* p = base + (size_t)(row0 + (lane & 15)) * ld + k0 + (lane >> 4) * 8;
  v16bf v;
#pragma unroll
  for (int i = 0; i < 8; ++i) v[i] = (bf16)p[i];
#pragma unroll
  for (int i = 0; i < 8; ++i) v[8 + i] = (bf16)p[16 + i];
  return v;
}

// B tile 32(K)x16(N) from row-major BT[N][K] (pre-transposed weights).
// ISA layout: lanes hold N=lane&15; elems 0-7 -> K = (lane>>4)*8 + 0..7,
// elems 8-15 -> K = 16 + (lane>>4)*8 + 0..7  => two 16B loads per lane.
__device__ __forceinline__ v16bf load_tileB_bf16(const bf16* base, int ld,
                                                 int col0, int k0) {
  int lane = threadIdx.x & 31;
  const bf16* p = base + (size_t)(col0 + (lane & 15)) * ld + k0 + (lane >> 4) * 8;
  union { v16bf v; uint4 q[2]; } u;
  u.q[0] = *(const uint4*)p;
  u.q[1] = *(const uint4*)(p + 16);
  return u.v;
}

// ---------------------------------------------------------------------------
// fp32 [R,C] -> bf16 [C,R] (batched via blockIdx.z). All dims here are 1024.
__global__ __launch_bounds__(256) void transpose_cvt_kernel(
    const float* __restrict__ src, bf16* __restrict__ dst, int R, int C) {
  __shared__ float tile[32][33];
  size_t bo = (size_t)blockIdx.z * R * C;
  src += bo; dst += bo;
  int c0 = blockIdx.x * 32, r0 = blockIdx.y * 32;
  int tx = threadIdx.x, ty = threadIdx.y;   // (32, 8)
#pragma unroll
  for (int i = 0; i < 32; i += 8)
    tile[ty + i][tx] = src[(size_t)(r0 + ty + i) * C + c0 + tx];
  __syncthreads();
#pragma unroll
  for (int i = 0; i < 32; i += 8)
    dst[(size_t)(c0 + ty + i) * R + r0 + tx] = (bf16)tile[tx][ty + i];
}

// w_gate [T,H,E] fp32 -> wgt [T*E][H] bf16 (tiny: 18K elements)
__global__ void wgate_cvt_kernel(const float* __restrict__ wg,
                                 bf16* __restrict__ wgt) {
  int i = blockIdx.x * 256 + threadIdx.x;
  if (i < NTASK * NEXP * DIMH) {
    int te = i / DIMH, k = i % DIMH;
    int t = te / NEXP, e = te % NEXP;
    wgt[i] = (bf16)wg[((size_t)t * DIMH + k) * NEXP + e];
  }
}

// ---------------------------------------------------------------------------
// C[M,N](bf16) = act(A[M,K] * BT[N,K]^T + bias). 256 thr = 8 waves,
// wave tile 32x32 (2x2 WMMA tiles), WG tile 64(M) x 128(N).
template <bool AF32, bool RELU>
__global__ __launch_bounds__(256) void gemm_bias_act(
    const void* __restrict__ Ap, int lda, const bf16* __restrict__ BT, int ldb,
    const float* __restrict__ bias, bf16* __restrict__ C, int ldc, int K) {
  int wave = threadIdx.x >> 5;
  int m0 = blockIdx.x * 64 + (wave & 1) * 32;
  int n0 = blockIdx.y * 128 + (wave >> 1) * 32;
  v8f acc[2][2];
#pragma unroll
  for (int i = 0; i < 2; ++i)
#pragma unroll
    for (int j = 0; j < 2; ++j) acc[i][j] = vzero8();

  for (int k0 = 0; k0 < K; k0 += 32) {
    v16bf a0, a1;
    if (AF32) {
      a0 = load_tileA_f32((const float*)Ap, lda, m0, k0);
      a1 = load_tileA_f32((const float*)Ap, lda, m0 + 16, k0);
    } else {
      a0 = load_tileA_bf16((const bf16*)Ap, lda, m0, k0);
      a1 = load_tileA_bf16((const bf16*)Ap, lda, m0 + 16, k0);
    }
    v16bf b0 = load_tileB_bf16(BT, ldb, n0, k0);
    v16bf b1 = load_tileB_bf16(BT, ldb, n0 + 16, k0);
    acc[0][0] = wmma_bf16(a0, b0, acc[0][0]);
    acc[0][1] = wmma_bf16(a0, b1, acc[0][1]);
    acc[1][0] = wmma_bf16(a1, b0, acc[1][0]);
    acc[1][1] = wmma_bf16(a1, b1, acc[1][1]);
  }

  int lane = threadIdx.x & 31, c = lane & 15, hi = lane >> 4;
#pragma unroll
  for (int i = 0; i < 2; ++i)
#pragma unroll
    for (int j = 0; j < 2; ++j) {
      int col = n0 + j * 16 + c;
      float bb = bias ? bias[col] : 0.f;
#pragma unroll
      for (int r = 0; r < 8; ++r) {
        float v = acc[i][j][r] + bb;
        if (RELU) v = fmaxf(v, 0.f);
        C[(size_t)(m0 + i * 16 + 8 * hi + r) * ldc + col] = (bf16)v;
      }
    }
}

// ---------------------------------------------------------------------------
// Fused MoE: one WG per 16-token tile. 512 thr = 16 waves, each wave owns a
// 64-wide column slab (4 WMMA tiles). Gating computed in fp32 into LDS; eh
// staged in LDS per expert; eo + gate combine kept in registers.
__global__ __launch_bounds__(512) void moe_expert_kernel(
    const bf16* __restrict__ hb, const bf16* __restrict__ wgt,
    const bf16* __restrict__ ew1t, const float* __restrict__ eb1,
    const bf16* __restrict__ ew2t, const float* __restrict__ eb2,
    bf16* __restrict__ mob) {
  __shared__ bf16 ehT[16][DIMH];                 // 32 KB
  __shared__ float logitsL[16][NTASK][NEXP];
  __shared__ float gatesL[NTASK][16][NEXP];

  const int tid = threadIdx.x;
  const int n0 = blockIdx.x * 16;

  // ---- gating: logits[row][t][e] = h[row] . wgt[t*E+e]  (bf16 in, f32 acc)
  if (tid < 16 * NTASK * NEXP) {
    int row = tid / (NTASK * NEXP);
    int te = tid % (NTASK * NEXP);
    const bf16* hrow = hb + (size_t)(n0 + row) * DIMH;
    const bf16* wrow = wgt + (size_t)te * DIMH;
    float s = 0.f;
    for (int k = 0; k < DIMH; k += 8) {
      union { uint4 q; bf16 h[8]; } ua, ub;
      ua.q = *(const uint4*)(hrow + k);
      ub.q = *(const uint4*)(wrow + k);
#pragma unroll
      for (int i = 0; i < 8; ++i) s += (float)ua.h[i] * (float)ub.h[i];
    }
    logitsL[row][te / NEXP][te % NEXP] = s;
  }
  __syncthreads();
  // ---- top-3 of 6 + softmax (first-index wins ties, like lax.top_k) ----
  if (tid < 16 * NTASK) {
    int row = tid / NTASK, t = tid % NTASK;
    float v[NEXP]; int used[NEXP];
    for (int e = 0; e < NEXP; ++e) { v[e] = logitsL[row][t][e]; used[e] = 0; }
    float tv[3]; int ti[3];
    for (int j = 0; j < 3; ++j) {
      float best = -3.4e38f; int bi = 0;
      for (int e = 0; e < NEXP; ++e)
        if (!used[e] && v[e] > best) { best = v[e]; bi = e; }
      used[bi] = 1; tv[j] = best; ti[j] = bi;
    }
    float w1 = __expf(tv[1] - tv[0]), w2 = __expf(tv[2] - tv[0]);
    float inv = 1.f / (1.f + w1 + w2);
    for (int e = 0; e < NEXP; ++e) gatesL[t][row][e] = 0.f;
    gatesL[t][row][ti[0]] = inv;
    gatesL[t][row][ti[1]] = w1 * inv;
    gatesL[t][row][ti[2]] = w2 * inv;
  }
  __syncthreads();

  const int wave = tid >> 5;
  const int lane = tid & 31, c = lane & 15, hi = lane >> 4;
  const int colBase = wave * 64;        // 16 waves * 64 = 1024 columns

  v8f mo[NTASK][4];
#pragma unroll
  for (int t = 0; t < NTASK; ++t)
#pragma unroll
    for (int j = 0; j < 4; ++j) mo[t][j] = vzero8();

  for (int e = 0; e < NEXP; ++e) {
    const bf16* w1 = ew1t + (size_t)e * DIMH * DIMH;   // [EH][H]
    const bf16* w2 = ew2t + (size_t)e * DIMH * DIMH;   // [H][EH]

    // ---- stage 1: eh = relu(h @ ew1 + eb1) -> LDS ----
    v8f acc[4];
#pragma unroll
    for (int j = 0; j < 4; ++j) acc[j] = vzero8();
    for (int k0 = 0; k0 < DIMH; k0 += 32) {
      v16bf a = load_tileA_bf16(hb, DIMH, n0, k0);     // hot in WGP$/L2
#pragma unroll
      for (int j = 0; j < 4; ++j) {
        v16bf b = load_tileB_bf16(w1, DIMH, colBase + j * 16, k0);
        acc[j] = wmma_bf16(a, b, acc[j]);
      }
    }
#pragma unroll
    for (int j = 0; j < 4; ++j) {
      int col = colBase + j * 16 + c;
      float bb = eb1[e * DIMH + col];
#pragma unroll
      for (int r = 0; r < 8; ++r)
        ehT[8 * hi + r][col] = (bf16)fmaxf(acc[j][r] + bb, 0.f);
    }
    __syncthreads();

    // ---- stage 2: eo = eh @ ew2 + eb2; mo[t] += gate[t,row,e] * eo ----
    v8f acc2[4];
#pragma unroll
    for (int j = 0; j < 4; ++j) acc2[j] = vzero8();
    for (int k0 = 0; k0 < DIMH; k0 += 32) {
      v16bf a = load_tileA_bf16(&ehT[0][0], DIMH, 0, k0);   // LDS reads
#pragma unroll
      for (int j = 0; j < 4; ++j) {
        v16bf b = load_tileB_bf16(w2, DIMH, colBase + j * 16, k0);
        acc2[j] = wmma_bf16(a, b, acc2[j]);
      }
    }
    float g[NTASK][8];
#pragma unroll
    for (int t = 0; t < NTASK; ++t)
#pragma unroll
      for (int r = 0; r < 8; ++r) g[t][r] = gatesL[t][8 * hi + r][e];
#pragma unroll
    for (int j = 0; j < 4; ++j) {
      float bb = eb2[e * DIMH + colBase + j * 16 + c];
#pragma unroll
      for (int r = 0; r < 8; ++r) {
        float eo = acc2[j][r] + bb;
#pragma unroll
        for (int t = 0; t < NTASK; ++t) mo[t][j][r] += g[t][r] * eo;
      }
    }
    __syncthreads();   // before next expert overwrites ehT
  }

  // ---- write mo bf16 [T][N][H] ----
#pragma unroll
  for (int t = 0; t < NTASK; ++t)
#pragma unroll
    for (int j = 0; j < 4; ++j) {
      int col = colBase + j * 16 + c;
#pragma unroll
      for (int r = 0; r < 8; ++r)
        mob[((size_t)t * NTOK + n0 + 8 * hi + r) * DIMH + col] =
            (bf16)mo[t][j][r];
    }
}

// ---------------------------------------------------------------------------
// Towers: th = relu(mo[t] @ tw1[t] + tb1[t]); out[t,n] += th . tw2[t]
// tw2-dot fused in epilogue: per-16-lane shfl_xor reduce + f32 atomics.
__global__ __launch_bounds__(256) void tower_kernel(
    const bf16* __restrict__ mob, const bf16* __restrict__ tw1t,
    const float* __restrict__ tb1, const float* __restrict__ tw2,
    float* __restrict__ out) {
  const int t = blockIdx.z;
  const bf16* A = mob + (size_t)t * NTOK * DIMH;
  const bf16* BT = tw1t + (size_t)t * DIMH * DIMH;   // [TH][H]
  const float* b1 = tb1 + t * DIMH;
  const float* w2 = tw2 + t * DIMH;

  int wave = threadIdx.x >> 5;
  int m0 = blockIdx.x * 64 + (wave & 1) * 32;
  int n0 = blockIdx.y * 128 + (wave >> 1) * 32;
  v8f acc[2][2];
#pragma unroll
  for (int i = 0; i < 2; ++i)
#pragma unroll
    for (int j = 0; j < 2; ++j) acc[i][j] = vzero8();

  for (int k0 = 0; k0 < DIMH; k0 += 32) {
    v16bf a0 = load_tileA_bf16(A, DIMH, m0, k0);
    v16bf a1 = load_tileA_bf16(A, DIMH, m0 + 16, k0);
    v16bf b0 = load_tileB_bf16(BT, DIMH, n0, k0);
    v16bf b1t = load_tileB_bf16(BT, DIMH, n0 + 16, k0);
    acc[0][0] = wmma_bf16(a0, b0, acc[0][0]);
    acc[0][1] = wmma_bf16(a0, b1t, acc[0][1]);
    acc[1][0] = wmma_bf16(a1, b0, acc[1][0]);
    acc[1][1] = wmma_bf16(a1, b1t, acc[1][1]);
  }

  int lane = threadIdx.x & 31, c = lane & 15, hi = lane >> 4;
#pragma unroll
  for (int i = 0; i < 2; ++i) {
    float p[8];
#pragma unroll
    for (int r = 0; r < 8; ++r) p[r] = 0.f;
#pragma unroll
    for (int j = 0; j < 2; ++j) {
      int col = n0 + j * 16 + c;
      float bb = b1[col], w = w2[col];
#pragma unroll
      for (int r = 0; r < 8; ++r)
        p[r] += fmaxf(acc[i][j][r] + bb, 0.f) * w;
    }
    // reduce over 16 columns (lanes 0-15 hold rows m0+i*16+r, 16-31 rows +8)
#pragma unroll
    for (int mask = 1; mask < 16; mask <<= 1)
#pragma unroll
      for (int r = 0; r < 8; ++r) p[r] += __shfl_xor(p[r], mask, 32);
    float myv = 0.f;
#pragma unroll
    for (int r = 0; r < 8; ++r) myv = (c == r) ? p[r] : myv;
    if (c < 8)
      atomicAdd(out + (size_t)t * NTOK + m0 + i * 16 + 8 * hi + c, myv);
  }
}

__global__ void out_init_kernel(float* __restrict__ out,
                                const float* __restrict__ tb2) {
  int i = blockIdx.x * 256 + threadIdx.x;
  if (i < NTASK * NTOK) out[i] = tb2[i / NTOK];
}

// ---------------------------------------------------------------------------
extern "C" void kernel_launch(void* const* d_in, const int* in_sizes, int n_in,
                              void* d_out, int out_size, void* d_ws,
                              size_t ws_size, hipStream_t stream) {
  (void)in_sizes; (void)n_in; (void)out_size; (void)ws_size;
  const float* x      = (const float*)d_in[0];   // [N, D]
  const float* fc1_w  = (const float*)d_in[1];   // [D, H]
  const float* fc1_b  = (const float*)d_in[2];
  const float* fc2_w  = (const float*)d_in[3];   // [H, H]
  const float* fc2_b  = (const float*)d_in[4];
  const float* w_gate = (const float*)d_in[5];   // [T, H, E]
  const float* ew1    = (const float*)d_in[6];   // [E, H, EH]
  const float* eb1    = (const float*)d_in[7];
  const float* ew2    = (const float*)d_in[8];   // [E, EH, H]
  const float* eb2    = (const float*)d_in[9];
  const float* tw1    = (const float*)d_in[10];  // [T, H, TH]
  const float* tb1    = (const float*)d_in[11];
  const float* tw2    = (const float*)d_in[12];  // [T, TH]
  const float* tb2    = (const float*)d_in[13];
  float* out = (float*)d_out;                    // [T, N]

  char* ws = (char*)d_ws;
  size_t off = 0;
  auto take = [&](size_t bytes) -> char* {
    char* p = ws + off;
    off += (bytes + 255) & ~(size_t)255;
    return p;
  };
  bf16* w1t  = (bf16*)take((size_t)DIMH * DIMH * 2);          // [H][D]
  bf16* w2t  = (bf16*)take((size_t)DIMH * DIMH * 2);          // [H][H]
  bf16* ew1t = (bf16*)take((size_t)NEXP * DIMH * DIMH * 2);   // [E][EH][H]
  bf16* ew2t = (bf16*)take((size_t)NEXP * DIMH * DIMH * 2);   // [E][H][EH]
  bf16* tw1t = (bf16*)take((size_t)NTASK * DIMH * DIMH * 2);  // [T][TH][H]
  bf16* wgt  = (bf16*)take((size_t)NTASK * NEXP * DIMH * 2);  // [T*E][H]
  bf16* h1b  = (bf16*)take((size_t)NTOK * DIMH * 2);          // relu(fc1)
  bf16* hb   = (bf16*)take((size_t)NTOK * DIMH * 2);          // shared bottom
  bf16* mob  = (bf16*)take((size_t)NTASK * NTOK * DIMH * 2);  // gate-combined

  dim3 tpb(32, 8);
  transpose_cvt_kernel<<<dim3(32, 32, 1), tpb, 0, stream>>>(fc1_w, w1t, DIMH, DIMH);
  transpose_cvt_kernel<<<dim3(32, 32, 1), tpb, 0, stream>>>(fc2_w, w2t, DIMH, DIMH);
  transpose_cvt_kernel<<<dim3(32, 32, NEXP), tpb, 0, stream>>>(ew1, ew1t, DIMH, DIMH);
  transpose_cvt_kernel<<<dim3(32, 32, NEXP), tpb, 0, stream>>>(ew2, ew2t, DIMH, DIMH);
  transpose_cvt_kernel<<<dim3(32, 32, NTASK), tpb, 0, stream>>>(tw1, tw1t, DIMH, DIMH);
  wgate_cvt_kernel<<<(NTASK * NEXP * DIMH + 255) / 256, 256, 0, stream>>>(
      w_gate, wgt);

  gemm_bias_act<true, true><<<dim3(NTOK / 64, DIMH / 128), 256, 0, stream>>>(
      x, DIMH, w1t, DIMH, fc1_b, h1b, DIMH, DIMH);
  gemm_bias_act<false, false><<<dim3(NTOK / 64, DIMH / 128), 256, 0, stream>>>(
      h1b, DIMH, w2t, DIMH, fc2_b, hb, DIMH, DIMH);

  moe_expert_kernel<<<NTOK / 16, 512, 0, stream>>>(hb, wgt, ew1t, eb1,
                                                   ew2t, eb2, mob);

  out_init_kernel<<<(NTASK * NTOK + 255) / 256, 256, 0, stream>>>(out, tb2);
  tower_kernel<<<dim3(NTOK / 64, DIMH / 128, NTASK), 256, 0, stream>>>(
      mob, tw1t, tb1, tw2, out);
}
